// Dynamic_Conv2D_79499844649612
// MI455X (gfx1250) — matrix operand order repaired and checked by reading, not compiled, not run on hardware
//
#include <hip/hip_runtime.h>

typedef __attribute__((ext_vector_type(16))) _Float16 v16h;
typedef __attribute__((ext_vector_type(8)))  _Float16 v8h;
typedef __attribute__((ext_vector_type(8)))  float    v8f;

#define B_    16
#define C_    64
#define H_    160
#define W_    160
#define OUT_  64
#define KEXP_ 4
#define HID_  17
#define KTOT  576           // 9 taps * 64 channels, K order = (ky*3+kx)*64 + c
#define HW_   (H_*W_)

// ---------------- kernel 1: global average pool -> pooled[B][C] ----------------
__global__ __launch_bounds__(256) void pool_kernel(const float* __restrict__ x,
                                                   float* __restrict__ pooled) {
    int bc = blockIdx.x;                                   // b*64 + c
    const float4* src = (const float4*)(x + (size_t)bc * HW_);
    float s = 0.f;
    for (int i = threadIdx.x; i < HW_ / 4; i += 256) {
        float4 v = src[i];
        s += v.x + v.y + v.z + v.w;
    }
    __shared__ float red[256];
    red[threadIdx.x] = s;
    __syncthreads();
    for (int off = 128; off > 0; off >>= 1) {
        if (threadIdx.x < off) red[threadIdx.x] += red[threadIdx.x + off];
        __syncthreads();
    }
    if (threadIdx.x == 0) pooled[bc] = red[0] * (1.0f / (float)HW_);
}

// ---------------- kernel 2: MLP + softmax -> attn[B][K], aggB[B][OUT] ----------------
__global__ __launch_bounds__(256) void attn_kernel(const float* __restrict__ pooled,
                                                   const float* __restrict__ fc1,
                                                   const float* __restrict__ fc2,
                                                   const float* __restrict__ bias_p,
                                                   float* __restrict__ attn,
                                                   float* __restrict__ aggB) {
    int t = threadIdx.x;
    if (t < B_) {
        float hid[HID_];
        for (int h = 0; h < HID_; ++h) {
            float s = 0.f;
            for (int c = 0; c < C_; ++c) s += pooled[t * C_ + c] * fc1[h * C_ + c];
            hid[h] = s > 0.f ? s : 0.f;
        }
        float lg[KEXP_];
        float m = -1e30f;
        for (int k = 0; k < KEXP_; ++k) {
            float s = 0.f;
            for (int h = 0; h < HID_; ++h) s += hid[h] * fc2[k * HID_ + h];
            lg[k] = s * (1.0f / 34.0f);
            m = fmaxf(m, lg[k]);
        }
        float se = 0.f;
        for (int k = 0; k < KEXP_; ++k) { lg[k] = __expf(lg[k] - m); se += lg[k]; }
        for (int k = 0; k < KEXP_; ++k) attn[t * KEXP_ + k] = lg[k] / se;
    }
    __syncthreads();
    for (int idx = t; idx < B_ * OUT_; idx += 256) {
        int b = idx / OUT_, o = idx % OUT_;
        float s = 0.f;
        for (int k = 0; k < KEXP_; ++k) s += attn[b * KEXP_ + k] * bias_p[k * OUT_ + o];
        aggB[idx] = s;
    }
}

// ---------------- kernel 3: aggregate expert weights -> f16, K=(tap)*64+c ----------------
__global__ __launch_bounds__(256) void aggw_kernel(const float* __restrict__ attn,
                                                   const float* __restrict__ weight,
                                                   _Float16* __restrict__ aggW) {
    int bo = blockIdx.x;                                   // b*64 + o
    int b = bo / OUT_, o = bo % OUT_;
    float a0 = attn[b * KEXP_ + 0], a1 = attn[b * KEXP_ + 1];
    float a2 = attn[b * KEXP_ + 2], a3 = attn[b * KEXP_ + 3];
    const size_t kstride = (size_t)OUT_ * C_ * 9;          // expert stride in weight[]
    for (int e = threadIdx.x; e < KTOT; e += 256) {
        int tap = e >> 6;                                  // ky*3 + kx
        int c = e & 63;
        int i = tap / 3, j = tap % 3;
        size_t widx = (((size_t)o * C_ + c) * 3 + i) * 3 + j;
        float v = a0 * weight[widx] + a1 * weight[widx + kstride] +
                  a2 * weight[widx + 2 * kstride] + a3 * weight[widx + 3 * kstride];
        aggW[(size_t)bo * KTOT + e] = (_Float16)v;
    }
}

// ---------------- kernel 4: implicit-GEMM conv via WMMA f16 ----------------
// grid = (H, B); block = 256 threads (8 waves). Block computes out[b, 0:64, y, 0:160].
__global__ __launch_bounds__(256) void conv_kernel(const float* __restrict__ x,
                                                   const _Float16* __restrict__ aggW,
                                                   const float* __restrict__ aggB,
                                                   float* __restrict__ out) {
    __shared__ _Float16 As[OUT_ * KTOT];                   // 73728 B: A[o][k]
    __shared__ _Float16 Xs[3 * 162 * C_];                  // 62208 B: [ky][x+1][c]

    const int y    = blockIdx.x;
    const int b    = blockIdx.y;
    const int tid  = threadIdx.x;
    const int lane = tid & 31;
    const int wave = tid >> 5;

    // ---- stage A: per-sample f16 weights (uint4 copies, coalesced b128) ----
    {
        const uint4* src = (const uint4*)(aggW + (size_t)b * (OUT_ * KTOT));
        uint4* dst = (uint4*)As;
        for (int i = tid; i < (OUT_ * KTOT) / 8; i += 256) dst[i] = src[i];
    }
    // ---- stage X: 3 input rows, f32->f16, transposed to channel-innermost ----
    for (int idx = tid; idx < 3 * C_ * W_; idx += 256) {
        int w   = idx % W_;
        int rem = idx / W_;
        int c   = rem % C_;
        int ky  = rem / C_;
        int row = y + ky - 1;
        float v = 0.f;
        if ((unsigned)row < (unsigned)H_)
            v = x[(((size_t)b * C_ + c) * H_ + row) * W_ + w];
        Xs[(ky * 162 + (w + 1)) * C_ + c] = (_Float16)v;
    }
    for (int idx = tid; idx < 3 * C_; idx += 256) {        // zero W halo columns
        int ky = idx / C_, c = idx % C_;
        Xs[(ky * 162 + 0)   * C_ + c] = (_Float16)0.f;
        Xs[(ky * 162 + 161) * C_ + c] = (_Float16)0.f;
    }
    __syncthreads();

    const int otile   = (wave & 3) * 16;                   // 4 M-tiles of 16 out-channels
    const int nbase   = (wave >> 2) * 80;                  // 2 groups x 80 pixels
    const int halfsel = lane >> 4;                         // 0 | 1
    const int l15     = lane & 15;

    // init accumulators with aggregated bias (C-matrix: M = vgpr + 8*halfsel)
    float bias[8];
#pragma unroll
    for (int r = 0; r < 8; ++r)
        bias[r] = aggB[b * OUT_ + otile + halfsel * 8 + r];
    v8f acc[5];
#pragma unroll
    for (int t = 0; t < 5; ++t)
#pragma unroll
        for (int r = 0; r < 8; ++r) acc[t][r] = bias[r];

    // ---- K loop: 18 steps of 32; one A fragment reused by 5 WMMAs ----
    for (int k0 = 0; k0 < KTOT; k0 += 32) {
        // A 16x32 fragment: lane<16 holds K[ak..ak+16) contiguous
        union { v16h v; v8h h[2]; } a;
        const int arow = otile + l15;
        const int ak   = k0 + halfsel * 16;
        a.h[0] = *(const v8h*)&As[arow * KTOT + ak];
        a.h[1] = *(const v8h*)&As[arow * KTOT + ak + 8];

        const int tap = k0 >> 6;                           // 32-chunk stays in one tap
        const int ky  = tap / 3, kx = tap % 3;
        const int cb  = (k0 & 63) + halfsel * 16;          // 16 consecutive channels
        const int xrb = ky * 162 + kx;
#pragma unroll
        for (int t = 0; t < 5; ++t) {
            int px = nbase + t * 16 + l15;                 // B column = pixel
            v16h bb = *(const v16h*)&Xs[(xrb + px) * C_ + cb];
            acc[t] = __builtin_amdgcn_wmma_f32_16x16x32_f16(
                false, a.v, false, bb, (short)0, acc[t], false, false);
        }
    }

    // ---- store: lanes 0-15 -> consecutive pixels (coalesced b32) ----
#pragma unroll
    for (int t = 0; t < 5; ++t) {
        int px = nbase + t * 16 + l15;
        size_t base = (((size_t)b * OUT_ + otile + halfsel * 8) * H_ + y) * W_ + px;
#pragma unroll
        for (int r = 0; r < 8; ++r)
            out[base + (size_t)r * HW_] = acc[t][r];
    }
}

extern "C" void kernel_launch(void* const* d_in, const int* in_sizes, int n_in,
                              void* d_out, int out_size, void* d_ws, size_t ws_size,
                              hipStream_t stream) {
    (void)in_sizes; (void)n_in; (void)out_size; (void)ws_size;
    const float* x      = (const float*)d_in[0];
    const float* fc1    = (const float*)d_in[1];
    const float* fc2    = (const float*)d_in[2];
    const float* weight = (const float*)d_in[3];
    const float* bias_p = (const float*)d_in[4];
    float* out = (float*)d_out;

    char* ws = (char*)d_ws;
    float*    pooled = (float*)(ws + 0);          // 4 KB
    float*    attn   = (float*)(ws + 4096);       // 256 B
    float*    aggB   = (float*)(ws + 8192);       // 4 KB
    _Float16* aggW   = (_Float16*)(ws + 16384);   // 1,179,648 B

    pool_kernel<<<B_ * C_, 256, 0, stream>>>(x, pooled);
    attn_kernel<<<1, 256, 0, stream>>>(pooled, fc1, fc2, bias_p, attn, aggB);
    aggw_kernel<<<B_ * OUT_, 256, 0, stream>>>(attn, weight, aggW);
    conv_kernel<<<dim3(H_, B_), 256, 0, stream>>>(x, aggW, aggB, out);
}
